// STDN_75582834475677
// MI455X (gfx1250) — matrix operand order repaired and checked by reading, not compiled, not run on hardware
//
#include <hip/hip_runtime.h>
#include <hip/hip_bf16.h>
#include <math.h>

typedef _Float16 h16;
typedef __attribute__((ext_vector_type(16))) _Float16 v16h;
typedef __attribute__((ext_vector_type(8)))  _Float16 v8h;
typedef __attribute__((ext_vector_type(8)))  float    v8f;

#define BN_EPS 1e-5f

// ---------------------------------------------------------------------------
// f32 -> f16 convert (weights, I)
// ---------------------------------------------------------------------------
__global__ void cvt_f32_to_f16(const float* __restrict__ src, h16* __restrict__ dst, int n) {
  int i = blockIdx.x * blockDim.x + threadIdx.x;
  if (i < n) dst[i] = (h16)src[i];
}

// ---------------------------------------------------------------------------
// Xc = concat(X, TE, SE) along channels, converted to f16.  P rows x 192 cols.
// ---------------------------------------------------------------------------
__global__ void pack_xc(const float* __restrict__ X, const float* __restrict__ TE,
                        const float* __restrict__ SE, h16* __restrict__ out, int P) {
  int i = blockIdx.x * blockDim.x + threadIdx.x;
  if (i >= P * 192) return;
  int p = i / 192, c = i % 192;
  float v;
  if (c < 64)       v = X [(size_t)p * 64 + c];
  else if (c < 128) v = TE[(size_t)p * 64 + (c - 64)];
  else              v = SE[(size_t)p * 64 + (c - 128)];
  out[i] = (h16)v;
}

// ---------------------------------------------------------------------------
// FC (1x1 conv) + BatchNorm(eval) + ReLU as a WMMA GEMM, weights staged in LDS.
//   Y[p,o] = relu( (sum_c A[p,c]*W[o,c] + b[o] - rm[o]) * g[o]*rsqrt(rv[o]+eps) + beta[o] )
// A: [P,C] f16 row-major. W: [O,C] f16 row-major == exactly the B-operand N x K
// layout for v_wmma_f32_16x16x32_f16.
//
// Block = 4 waves. Whole W (<= 24 KB f16) is async-copied to LDS once per block
// (global_load_async_to_lds_b128 + s_wait_asynccnt + workgroup barrier), then
// each wave owns one 16-row M tile and ALL O/16 column tiles: the A fragment is
// loaded once per 32-wide K chunk and reused across O/16 back-to-back WMMAs.
// ---------------------------------------------------------------------------
template<int C, int O>
__global__ __launch_bounds__(128)
void fc_gemm_wmma(const h16* __restrict__ A, const h16* __restrict__ W,
                  const float* __restrict__ bias, const float* __restrict__ gain,
                  const float* __restrict__ beta, const float* __restrict__ rm,
                  const float* __restrict__ rv,
                  h16* __restrict__ Y16, float* __restrict__ Y32,
                  const float* __restrict__ Xadd,
                  int P)
{
  __shared__ __align__(64) h16 lw[O * C];

  // ---- async copy of whole weight matrix into LDS (ASYNCcnt-tracked) ------
  // 16 bytes per enabled lane per issue; O*C*2 is a multiple of 128*16.
  constexpr int TOTAL = O * C * 2;
#pragma unroll 1
  for (int off = threadIdx.x * 16; off < TOTAL; off += 128 * 16) {
    uint32_t la = (uint32_t)(uintptr_t)((char*)lw + off);            // addr[31:0] = LDS offset
    uint64_t ga = (uint64_t)(uintptr_t)((const char*)W + off);
    asm volatile("global_load_async_to_lds_b128 %0, %1, off"
                 :: "v"(la), "v"(ga) : "memory");
  }
  asm volatile("s_wait_asynccnt 0x0" ::: "memory");
  __syncthreads();

  const int lane = threadIdx.x & 31;
  const int wave = threadIdx.x >> 5;
  const int tm   = blockIdx.x * 4 + wave;
  if (tm >= (P >> 4)) return;              // wave-uniform exit (EXEC stays all-ones)

  // A operand: lanes 0-15 -> M=lane, K chunks {k..k+7, k+16..k+23};
  //            lanes 16-31 -> same M, K chunks {k+8..k+15, k+24..k+31}.
  const int rowA = tm * 16 + (lane & 15);
  const int hiA  = (lane >> 4) ? 8 : 0;
  // B operand: lanes 0-15 -> N=lane, K chunks {k..k+7, k+16..k+23};
  //            lanes 16-31 -> same N, K chunks {k+8..k+15, k+24..k+31}.
  const int hiW  = (lane >> 4) ? 8 : 0;

  const h16* aBase = A + (size_t)rowA * C;
  const h16* wRow  = lw + (size_t)(lane & 15) * C;   // + t*16*C selects column tile

  v8f acc[O / 16];
#pragma unroll
  for (int t = 0; t < O / 16; ++t) acc[t] = (v8f){};

#pragma unroll
  for (int k = 0; k < C; k += 32) {
    if (k + 32 < C) __builtin_prefetch(aBase + k + 32, 0, 3);        // global_prefetch_b8
    v8h a0 = *(const v8h*)(aBase + k + hiA);
    v8h a1 = *(const v8h*)(aBase + k + hiA + 16);
    v16h av;
#pragma unroll
    for (int i = 0; i < 8; ++i) { av[i] = a0[i]; av[i + 8] = a1[i]; }
#pragma unroll
    for (int t = 0; t < O / 16; ++t) {
      v8h b0 = *(const v8h*)(wRow + (size_t)t * 16 * C + k + hiW); // ds_load from LDS
      v8h b1 = *(const v8h*)(wRow + (size_t)t * 16 * C + k + hiW + 16);
      v16h bv;
#pragma unroll
      for (int i = 0; i < 8; ++i) { bv[i] = b0[i]; bv[i + 8] = b1[i]; }
      acc[t] = __builtin_amdgcn_wmma_f32_16x16x32_f16(false, av, false, bv,
                                                      (short)0, acc[t], false, false);
    }
  }

  // C/D layout: VGPR r -> M = tm*16 + r (+8 upper lanes), N (=channel o) = lane&15.
#pragma unroll
  for (int t = 0; t < O / 16; ++t) {
    const int o = t * 16 + (lane & 15);
    const float scale = gain[o] * rsqrtf(rv[o] + BN_EPS);
    const float shift = (bias[o] - rm[o]) * scale + beta[o];
#pragma unroll
    for (int r = 0; r < 8; ++r) {
      const int p = tm * 16 + r + hiA;
      float y = acc[t][r] * scale + shift;
      y = y > 0.f ? y : 0.f;
      if (Y16) Y16[(size_t)p * O + o] = (h16)y;
      if (Y32) {
        float base = Xadd ? Xadd[(size_t)p * O + o] : 0.f;
        Y32[(size_t)p * O + o] = base + y;
      }
    }
  }
}

// ---------------------------------------------------------------------------
// Per-(b,n) multi-head attention; heads=8, head_dim=8, tiny Tq/Tk -> VALU.
// QSHARED: q rows are batch-invariant (mab0's broadcast inducing points).
// One thread per (head, tq); blockDim = 8*TQ.
// ---------------------------------------------------------------------------
template<int TQ, int TK, bool QSHARED>
__global__ void attn_kernel(const h16* __restrict__ Qm, const h16* __restrict__ Km,
                            const h16* __restrict__ Vm, h16* __restrict__ out,
                            int Bb, int Nn)
{
  const int bn = blockIdx.x;
  const int b  = bn / Nn;
  const int n  = bn % Nn;
  const int h  = threadIdx.x / TQ;   // 0..7
  const int tq = threadIdx.x % TQ;

  const size_t qrow = QSHARED ? ((size_t)(tq * Nn + n))
                              : ((size_t)((b * TQ + tq) * Nn + n));
  const h16* qr = Qm + qrow * 64 + h * 8;
  float q[8];
#pragma unroll
  for (int c = 0; c < 8; ++c) q[c] = (float)qr[c];

  const float sc = 0.35355339059327373f;   // 1/sqrt(d_att=8)
  float s[TK];
  float mx = -1e30f;
  for (int tk = 0; tk < TK; ++tk) {
    const h16* kr = Km + ((size_t)((b * TK + tk) * Nn + n)) * 64 + h * 8;
    float d = 0.f;
#pragma unroll
    for (int c = 0; c < 8; ++c) d += q[c] * (float)kr[c];
    d *= sc;
    s[tk] = d;
    mx = fmaxf(mx, d);
  }
  float sum = 0.f;
  for (int tk = 0; tk < TK; ++tk) { s[tk] = __expf(s[tk] - mx); sum += s[tk]; }
  const float inv = 1.f / sum;

  float o8[8] = {0.f, 0.f, 0.f, 0.f, 0.f, 0.f, 0.f, 0.f};
  for (int tk = 0; tk < TK; ++tk) {
    const h16* vr = Vm + ((size_t)((b * TK + tk) * Nn + n)) * 64 + h * 8;
    const float w = s[tk] * inv;
#pragma unroll
    for (int c = 0; c < 8; ++c) o8[c] += w * (float)vr[c];
  }
  h16* orow = out + ((size_t)((b * TQ + tq) * Nn + n)) * 64 + h * 8;
#pragma unroll
  for (int c = 0; c < 8; ++c) orow[c] = (h16)o8[c];
}

// ---------------------------------------------------------------------------
// Host side
// ---------------------------------------------------------------------------
struct Fc { const float *W, *b, *g, *beta, *rm, *rv; };
static Fc get_fc(void* const* d_in, int base) {
  Fc f;
  f.W    = (const float*)d_in[base + 0];
  f.b    = (const float*)d_in[base + 1];
  f.g    = (const float*)d_in[base + 2];
  f.beta = (const float*)d_in[base + 3];
  f.rm   = (const float*)d_in[base + 4];
  f.rv   = (const float*)d_in[base + 5];
  return f;
}

extern "C" void kernel_launch(void* const* d_in, const int* in_sizes, int n_in,
                              void* d_out, int out_size, void* d_ws, size_t ws_size,
                              hipStream_t stream)
{
  (void)in_sizes; (void)n_in; (void)out_size; (void)ws_size;

  const int Bb = 64, T = 12, N = 307, SET = 16;
  const int P  = Bb * T * N;     // 235776 (divisible by 16)
  const int Qr = Bb * SET * N;   // 314368 (divisible by 16)
  const int PI = SET * N;        // 4912   (divisible by 16)

  // Inputs in setup_inputs() dict insertion order:
  // X, TE, SE, params.I, params.mab0{q,k,v,o}, params.mab1{q,k,v,o} (each: W,b,g,beta,rm,rv), mask
  const float* X  = (const float*)d_in[0];
  const float* TE = (const float*)d_in[1];
  const float* SE = (const float*)d_in[2];
  const float* Ip = (const float*)d_in[3];
  Fc m0q = get_fc(d_in,  4);
  Fc m0k = get_fc(d_in, 10);
  Fc m0v = get_fc(d_in, 16);
  Fc m0o = get_fc(d_in, 22);
  Fc m1q = get_fc(d_in, 28);
  Fc m1k = get_fc(d_in, 34);
  Fc m1v = get_fc(d_in, 40);
  Fc m1o = get_fc(d_in, 46);

  // Workspace layout (all f16 intermediates to halve HBM traffic; 256B aligned)
  char* ws = (char*)d_ws;
  size_t off = 0;
  auto alloc = [&](size_t bytes) -> char* {
    char* p = ws + off;
    off += (bytes + 255) & ~(size_t)255;
    return p;
  };
  h16* I16    = (h16*)alloc((size_t)PI * 192 * 2);
  h16* Xc16   = (h16*)alloc((size_t)P  * 192 * 2);
  h16* wq0    = (h16*)alloc(64 * 192 * 2);
  h16* wk0    = (h16*)alloc(64 * 192 * 2);
  h16* wv0    = (h16*)alloc(64 * 192 * 2);
  h16* wo0    = (h16*)alloc(192 * 64 * 2);
  h16* wq1    = (h16*)alloc(64 * 192 * 2);
  h16* wk1    = (h16*)alloc(64 * 192 * 2);
  h16* wv1    = (h16*)alloc(64 * 192 * 2);
  h16* wo1    = (h16*)alloc(64 * 64 * 2);
  h16* q016   = (h16*)alloc((size_t)PI * 64 * 2);   // batch-invariant q of mab0
  h16* k016   = (h16*)alloc((size_t)P  * 64 * 2);
  h16* v016   = (h16*)alloc((size_t)P  * 64 * 2);
  h16* q116   = (h16*)alloc((size_t)P  * 64 * 2);
  h16* at016  = (h16*)alloc((size_t)Qr * 64 * 2);
  h16* H16    = (h16*)alloc((size_t)Qr * 192 * 2);
  h16* k116   = (h16*)alloc((size_t)Qr * 64 * 2);
  h16* v116   = (h16*)alloc((size_t)Qr * 64 * 2);
  h16* at116  = (h16*)alloc((size_t)P  * 64 * 2);

  auto cvt = [&](const float* s, h16* d, int n) {
    cvt_f32_to_f16<<<(n + 255) / 256, 256, 0, stream>>>(s, d, n);
  };
  cvt(Ip,     I16, PI * 192);
  cvt(m0q.W,  wq0, 64 * 192);
  cvt(m0k.W,  wk0, 64 * 192);
  cvt(m0v.W,  wv0, 64 * 192);
  cvt(m0o.W,  wo0, 192 * 64);
  cvt(m1q.W,  wq1, 64 * 192);
  cvt(m1k.W,  wk1, 64 * 192);
  cvt(m1v.W,  wv1, 64 * 192);
  cvt(m1o.W,  wo1, 64 * 64);

  {
    int n = P * 192;
    pack_xc<<<(n + 255) / 256, 256, 0, stream>>>(X, TE, SE, Xc16, P);
  }

#define GEMM(CC, OO, Aptr, Wptr, f, y16, y32, xadd, Pn)                          \
  do {                                                                           \
    int blocks_ = (((Pn) / 16) + 3) / 4;                                         \
    fc_gemm_wmma<CC, OO><<<blocks_, 128, 0, stream>>>(                           \
        (Aptr), (Wptr), (f).b, (f).g, (f).beta, (f).rm, (f).rv,                  \
        (y16), (y32), (xadd), (Pn));                                             \
  } while (0)

  // MAB0
  GEMM(192, 64, I16,  wq0, m0q, q016, (float*)nullptr, (const float*)nullptr, PI); // q (batch-invariant)
  GEMM(192, 64, Xc16, wk0, m0k, k016, (float*)nullptr, (const float*)nullptr, P);  // k
  GEMM(192, 64, Xc16, wv0, m0v, v016, (float*)nullptr, (const float*)nullptr, P);  // v
  GEMM(192, 64, Xc16, wq1, m1q, q116, (float*)nullptr, (const float*)nullptr, P);  // mab1 q (from Xc)
  attn_kernel<16, 12, true><<<Bb * N, 128, 0, stream>>>(q016, k016, v016, at016, Bb, N);
  GEMM(64, 192, at016, wo0, m0o, H16, (float*)nullptr, (const float*)nullptr, Qr); // o -> H

  // MAB1
  GEMM(192, 64, H16, wk1, m1k, k116, (float*)nullptr, (const float*)nullptr, Qr);
  GEMM(192, 64, H16, wv1, m1v, v116, (float*)nullptr, (const float*)nullptr, Qr);
  attn_kernel<12, 16, false><<<Bb * N, 96, 0, stream>>>(q116, k116, v116, at116, Bb, N);
  // Final FC with fused residual: d_out = X + relu(bn(at1 @ Wo1))
  GEMM(64, 64, at116, wo1, m1o, (h16*)nullptr, (float*)d_out, X, P);

#undef GEMM
}
